// CrossAttentionCLSHead_39281770889812
// MI455X (gfx1250) — matrix operand order repaired and checked by reading, not compiled, not run on hardware
//
#include <hip/hip_runtime.h>
#include <hip/hip_bf16.h>

typedef __bf16 bf16;
typedef __attribute__((ext_vector_type(16))) __bf16 v16bf;
typedef __attribute__((ext_vector_type(8)))  float   v8f;
typedef __attribute__((ext_vector_type(4)))  float   v4f;

constexpr int cB  = 8;
constexpr int cL  = 2048;
constexpr int cD  = 1024;
constexpr int cH  = 1024;
constexpr int cNH = 16;
constexpr int cHD = 64;
constexpr float cEPS = 1e-5f;

// ---------------------------------------------------------------------------
// f32 -> bf16 convert (weights)
// ---------------------------------------------------------------------------
__global__ __launch_bounds__(256) void cvt_bf16_kernel(const float* __restrict__ in,
                                                       bf16* __restrict__ out, long n) {
  long i = (long)blockIdx.x * 256 + threadIdx.x;
  if (i < n) out[i] = (bf16)in[i];
}

// ---------------------------------------------------------------------------
// Row LayerNorm (block per row), optional per-row mask multiply
// ---------------------------------------------------------------------------
__global__ __launch_bounds__(256) void ln_rows_kernel(const float* __restrict__ x,
                                                      const float* __restrict__ w,
                                                      const float* __restrict__ bb,
                                                      const int* __restrict__ mask,
                                                      float* __restrict__ out) {
  int row = blockIdx.x;
  const float* xr = x + (long)row * cH;
  float* orow = out + (long)row * cH;
  int tid = threadIdx.x;
  float s = 0.f, s2 = 0.f;
  for (int i = tid; i < cH; i += 256) { float v = xr[i]; s += v; s2 += v * v; }
  __shared__ float rs[256], rs2[256];
  rs[tid] = s; rs2[tid] = s2; __syncthreads();
  for (int off = 128; off > 0; off >>= 1) {
    if (tid < off) { rs[tid] += rs[tid + off]; rs2[tid] += rs2[tid + off]; }
    __syncthreads();
  }
  float mean = rs[0] * (1.0f / cH);
  float var  = rs2[0] * (1.0f / cH) - mean * mean;
  float inv  = rsqrtf(var + cEPS);
  float mval = mask ? (float)mask[row] : 1.0f;
  for (int i = tid; i < cH; i += 256) {
    float v = (xr[i] - mean) * inv * w[i] + bb[i];
    orow[i] = v * mval;
  }
}

// ---------------------------------------------------------------------------
// WMMA GEMM: C(MxN) = A(MxK, f32) @ W(NxK, bf16)^T + bias.
// Each wave computes a 16x64 strip (4 N-tiles): the f32 A fragment (load +
// bf16 pack) is amortized over 4 independent WMMAs per K-step.
// mode 0: f32 row-major out (scaled)   mode 1: bf16 row-major out
// mode 2: bf16 transposed out  vt[b, h, hd, seq]  (seqLen rows per batch)
// ---------------------------------------------------------------------------
__global__ __launch_bounds__(256) void gemm_wmma_kernel(const float* __restrict__ A,
                                                        const bf16* __restrict__ W,
                                                        const float* __restrict__ bias,
                                                        void* __restrict__ out,
                                                        int M, int N, int K,
                                                        float outScale, int mode, int seqLen) {
  int wave = threadIdx.x >> 5;
  int lane = threadIdx.x & 31;
  long wtile = (long)blockIdx.x * 8 + wave;
  int tilesN4 = N >> 6;                           // strips of 4 N-tiles (64 cols)
  long total = (long)(M >> 4) * tilesN4;
  if (wtile >= total) return;                     // wave-uniform
  int tm  = (int)(wtile / tilesN4);
  int tn4 = (int)(wtile % tilesN4);
  int row0 = tm << 4, col0 = tn4 << 6;
  int mrow = lane & 15;                           // A row / B col within tile
  int khalf = lane >> 4;

  const float* arow  = A + (long)(row0 + mrow) * K + khalf * 16;
  const bf16*  brow0 = W + (long)(col0 + mrow) * K + khalf * 16;
  const bf16*  brow1 = brow0 + (long)16 * K;
  const bf16*  brow2 = brow0 + (long)32 * K;
  const bf16*  brow3 = brow0 + (long)48 * K;

  v8f c0 = {}, c1 = {}, c2 = {}, c3 = {};
  for (int k0 = 0; k0 < K; k0 += 32) {
    v4f a0 = *(const v4f*)(arow + k0);
    v4f a1 = *(const v4f*)(arow + k0 + 4);
    v4f a2 = *(const v4f*)(arow + k0 + 8);
    v4f a3 = *(const v4f*)(arow + k0 + 12);
    v16bf a;
#pragma unroll
    for (int j = 0; j < 4; ++j) {
      a[j]      = (bf16)a0[j];
      a[4 + j]  = (bf16)a1[j];
      a[8 + j]  = (bf16)a2[j];
      a[12 + j] = (bf16)a3[j];
    }
    v16bf b0 = *(const v16bf*)(brow0 + k0);
    v16bf b1 = *(const v16bf*)(brow1 + k0);
    v16bf b2 = *(const v16bf*)(brow2 + k0);
    v16bf b3 = *(const v16bf*)(brow3 + k0);
    c0 = __builtin_amdgcn_wmma_f32_16x16x32_bf16(false, a, false, b0, (short)0, c0, false, false);
    c1 = __builtin_amdgcn_wmma_f32_16x16x32_bf16(false, a, false, b1, (short)0, c1, false, false);
    c2 = __builtin_amdgcn_wmma_f32_16x16x32_bf16(false, a, false, b2, (short)0, c2, false, false);
    c3 = __builtin_amdgcn_wmma_f32_16x16x32_bf16(false, a, false, b3, (short)0, c3, false, false);
  }

#pragma unroll
  for (int t = 0; t < 4; ++t) {
    v8f c = (t == 0) ? c0 : (t == 1) ? c1 : (t == 2) ? c2 : c3;
    int n = col0 + t * 16 + mrow;
    float bv = bias[n];
#pragma unroll
    for (int r = 0; r < 8; ++r) {
      int mG = row0 + r + khalf * 8;
      float val = (c[r] + bv) * outScale;
      if (mode == 0) {
        ((float*)out)[(long)mG * N + n] = val;
      } else if (mode == 1) {
        ((bf16*)out)[(long)mG * N + n] = (bf16)val;
      } else {
        int bIdx = mG / seqLen, s = mG % seqLen;
        int hh = n >> 6, hd = n & 63;             // HD = 64
        ((bf16*)out)[(((long)bIdx * cNH + hh) * cHD + hd) * seqLen + s] = (bf16)val;
      }
    }
  }
}

// ---------------------------------------------------------------------------
// Fused attention: block per (b, h, 16-query tile). Scores (WMMA) -> LDS,
// key-masked softmax, write attn (B,NH,D,L) and ctx (B,D,H) = P @ V (WMMA).
// ---------------------------------------------------------------------------
__global__ __launch_bounds__(256) void attn_fused_kernel(const float* __restrict__ qp,   // (B,D,H) pre-scaled
                                                         const bf16* __restrict__ kp,    // (B,L,NH,HD)
                                                         const bf16* __restrict__ vt,    // (B,NH,HD,L)
                                                         const int* __restrict__ dmask,  // (B,L)
                                                         float* __restrict__ attn,       // (B,NH,D,L)
                                                         float* __restrict__ ctx) {      // (B,D,H)
  extern __shared__ float smem[];
  float* sc     = smem;                 // 16 x 2048
  float* redm   = sc + 16 * cL;         // 16 x 16
  float* reds   = redm + 256;           // 16 x 16
  float* rowmax = reds + 256;           // 16
  float* rowsum = rowmax + 16;          // 16

  int blk = blockIdx.x;
  int qt = blk & 63;                    // D/16 = 64
  int h  = (blk >> 6) & (cNH - 1);
  int b  = blk >> 10;
  int q0 = qt << 4;
  int wave = threadIdx.x >> 5, lane = threadIdx.x & 31;
  int mrow = lane & 15, khalf = lane >> 4;

  // Q A-fragments for HD=64 (k-steps 0 and 32)
  const float* qrow = qp + ((long)(b * cD + q0 + mrow)) * cH + h * cHD + khalf * 16;
  v16bf aq0, aq1;
  {
    v4f t0 = *(const v4f*)(qrow + 0);
    v4f t1 = *(const v4f*)(qrow + 4);
    v4f t2 = *(const v4f*)(qrow + 8);
    v4f t3 = *(const v4f*)(qrow + 12);
    v4f u0 = *(const v4f*)(qrow + 32);
    v4f u1 = *(const v4f*)(qrow + 36);
    v4f u2 = *(const v4f*)(qrow + 40);
    v4f u3 = *(const v4f*)(qrow + 44);
#pragma unroll
    for (int j = 0; j < 4; ++j) {
      aq0[j]      = (bf16)t0[j]; aq0[4 + j]  = (bf16)t1[j];
      aq0[8 + j]  = (bf16)t2[j]; aq0[12 + j] = (bf16)t3[j];
      aq1[j]      = (bf16)u0[j]; aq1[4 + j]  = (bf16)u1[j];
      aq1[8 + j]  = (bf16)u2[j]; aq1[12 + j] = (bf16)u3[j];
    }
  }

  // scores: each wave handles 16 key tiles of 16 keys (Q fragment reused)
  for (int i = 0; i < 16; ++i) {
    int key0 = (wave * 16 + i) << 4;
    const bf16* krow = kp + ((long)(b * cL + key0 + mrow)) * cH + h * cHD + khalf * 16;
    v16bf kb0 = *(const v16bf*)(krow);
    v16bf kb1 = *(const v16bf*)(krow + 32);
    v8f c = {};
    c = __builtin_amdgcn_wmma_f32_16x16x32_bf16(false, aq0, false, kb0, (short)0, c, false, false);
    c = __builtin_amdgcn_wmma_f32_16x16x32_bf16(false, aq1, false, kb1, (short)0, c, false, false);
    int key = key0 + mrow;
    float mv = (dmask[b * cL + key] != 0) ? 0.0f : -3.0e38f;
    float* col = sc + key;
#pragma unroll
    for (int r = 0; r < 8; ++r) col[(r + khalf * 8) * cL] = c[r] + mv;
  }
  __syncthreads();

  // softmax: 16 threads per row
  int tid = threadIdx.x;
  int r = tid >> 4, sub = tid & 15;
  float* row = sc + r * cL;
  float lmax = -3.4e38f;
  for (int c0 = sub; c0 < cL; c0 += 16) lmax = fmaxf(lmax, row[c0]);
  redm[r * 16 + sub] = lmax; __syncthreads();
  if (sub == 0) {
    float m2 = redm[r * 16];
    for (int j = 1; j < 16; ++j) m2 = fmaxf(m2, redm[r * 16 + j]);
    rowmax[r] = m2;
  }
  __syncthreads();
  float mR = rowmax[r];
  float lsum = 0.f;
  for (int c0 = sub; c0 < cL; c0 += 16) {
    float e = __expf(row[c0] - mR);
    row[c0] = e; lsum += e;
  }
  reds[r * 16 + sub] = lsum; __syncthreads();
  if (sub == 0) {
    float s2 = 0.f;
    for (int j = 0; j < 16; ++j) s2 += reds[r * 16 + j];
    rowsum[r] = s2;
  }
  __syncthreads();
  float inv = 1.0f / rowsum[r];
  float* arow = attn + (((long)(b * cNH + h)) * cD + q0 + r) * cL;
  for (int c0 = sub; c0 < cL; c0 += 16) {
    float p = row[c0] * inv;
    row[c0] = p;
    arow[c0] = p;
  }
  __syncthreads();

  // ctx = P(16xL) @ V(LxHD): waves 0..3 each produce a 16x16 hd-tile
  if (wave < 4) {
    int n0 = wave << 4;
    const bf16* vrow = vt + (((long)(b * cNH + h)) * cHD + n0 + mrow) * cL + khalf * 16;
    v8f c = {};
    for (int k0 = 0; k0 < cL; k0 += 32) {
      const float* sp = sc + mrow * cL + k0 + khalf * 16;
      v4f a0 = *(const v4f*)(sp);
      v4f a1 = *(const v4f*)(sp + 4);
      v4f a2 = *(const v4f*)(sp + 8);
      v4f a3 = *(const v4f*)(sp + 12);
      v16bf a;
#pragma unroll
      for (int j = 0; j < 4; ++j) {
        a[j]      = (bf16)a0[j];
        a[4 + j]  = (bf16)a1[j];
        a[8 + j]  = (bf16)a2[j];
        a[12 + j] = (bf16)a3[j];
      }
      v16bf bv = *(const v16bf*)(vrow + k0);
      c = __builtin_amdgcn_wmma_f32_16x16x32_bf16(false, a, false, bv, (short)0, c, false, false);
    }
#pragma unroll
    for (int rr = 0; rr < 8; ++rr) {
      int m = rr + khalf * 8;
      ctx[((long)(b * cD + q0 + m)) * cH + h * cHD + n0 + mrow] = c[rr];
    }
  }
}

// ---------------------------------------------------------------------------
// zero pooled + per-batch valid-query count
// ---------------------------------------------------------------------------
__global__ __launch_bounds__(256) void zero_pool_count_kernel(float* __restrict__ pooled,
                                                              const int* __restrict__ dmask,
                                                              float* __restrict__ cnt) {
  int b = blockIdx.x, tid = threadIdx.x;
  for (int i = tid; i < cH; i += 256) pooled[b * cH + i] = 0.f;
  int c = 0;
  for (int i = tid; i < cD; i += 256) c += dmask[b * cD + i];
  __shared__ int rc[256];
  rc[tid] = c; __syncthreads();
  for (int off = 128; off > 0; off >>= 1) {
    if (tid < off) rc[tid] += rc[tid + off];
    __syncthreads();
  }
  if (tid == 0) cnt[b] = fmaxf((float)rc[0], 1e-8f);
}

// ---------------------------------------------------------------------------
// final: x = out_proj(ctx)+q residual -> LN -> mask -> pooled accumulation
// ---------------------------------------------------------------------------
__global__ __launch_bounds__(256) void final_pool_kernel(const float* __restrict__ ao,
                                                         const float* __restrict__ qn,
                                                         const float* __restrict__ w,
                                                         const float* __restrict__ bb,
                                                         const int* __restrict__ dmask,
                                                         const float* __restrict__ cnt,
                                                         float* __restrict__ pooled) {
  int rowIdx = blockIdx.x;            // b*D + d
  int b = rowIdx >> 10;               // D = 1024
  if (dmask[rowIdx] == 0) return;     // masked rows contribute zero
  const float* xr = ao + (long)rowIdx * cH;
  const float* qr = qn + (long)rowIdx * cH;
  int tid = threadIdx.x;
  __shared__ float xbuf[cH];
  float s = 0.f, s2 = 0.f;
  for (int i = tid; i < cH; i += 256) {
    float v = xr[i] + qr[i];
    xbuf[i] = v; s += v; s2 += v * v;
  }
  __shared__ float rs[256], rs2[256];
  rs[tid] = s; rs2[tid] = s2; __syncthreads();
  for (int off = 128; off > 0; off >>= 1) {
    if (tid < off) { rs[tid] += rs[tid + off]; rs2[tid] += rs2[tid + off]; }
    __syncthreads();
  }
  float mean = rs[0] * (1.0f / cH);
  float var  = rs2[0] * (1.0f / cH) - mean * mean;
  float inv  = rsqrtf(var + cEPS);
  float invc = 1.0f / cnt[b];
  for (int i = tid; i < cH; i += 256) {
    float y = (xbuf[i] - mean) * inv * w[i] + bb[i];
    atomicAdd(&pooled[b * cH + i], y * invc);
  }
}

// ---------------------------------------------------------------------------
extern "C" void kernel_launch(void* const* d_in, const int* in_sizes, int n_in,
                              void* d_out, int out_size, void* d_ws, size_t ws_size,
                              hipStream_t stream) {
  const float* dna       = (const float*)d_in[0];
  const float* desc      = (const float*)d_in[1];
  const int*   dna_mask  = (const int*)d_in[2];
  const int*   desc_mask = (const int*)d_in[3];
  const float* qw  = (const float*)d_in[4];
  const float* qb  = (const float*)d_in[5];
  const float* kvw = (const float*)d_in[6];
  const float* kvb = (const float*)d_in[7];
  const float* inw = (const float*)d_in[8];
  const float* inb = (const float*)d_in[9];
  const float* ow  = (const float*)d_in[10];
  const float* ob  = (const float*)d_in[11];
  const float* onw = (const float*)d_in[12];
  const float* onb = (const float*)d_in[13];

  char* ws = (char*)d_ws;
  auto carve = [&](size_t bytes) {
    char* p = ws;
    ws += (bytes + 255) & ~(size_t)255;
    return p;
  };
  bf16*  wbf = (bf16*)carve((size_t)4 * cH * cH * sizeof(bf16));   // Wq|Wk|Wv|Wo
  float* qn  = (float*)carve((size_t)cB * cD * cH * sizeof(float));
  float* kvn = (float*)carve((size_t)cB * cL * cH * sizeof(float));
  float* qp  = (float*)carve((size_t)cB * cD * cH * sizeof(float));
  bf16*  kp  = (bf16*)carve((size_t)cB * cL * cH * sizeof(bf16));
  bf16*  vt  = (bf16*)carve((size_t)cB * cL * cH * sizeof(bf16));
  float* ctx = (float*)carve((size_t)cB * cD * cH * sizeof(float));
  float* ao  = (float*)carve((size_t)cB * cD * cH * sizeof(float));
  float* cnt = (float*)carve((size_t)cB * sizeof(float));

  float* pooled = (float*)d_out;                 // (B,H)
  float* attn   = (float*)d_out + cB * cH;       // (B,NH,D,L)

  // 1) weights -> bf16
  long n_in_w = (long)3 * cH * cH;
  cvt_bf16_kernel<<<(int)((n_in_w + 255) / 256), 256, 0, stream>>>(inw, wbf, n_in_w);
  long n_out_w = (long)cH * cH;
  cvt_bf16_kernel<<<(int)((n_out_w + 255) / 256), 256, 0, stream>>>(ow, wbf + 3L * cH * cH, n_out_w);

  // 2) LayerNorms
  ln_rows_kernel<<<cB * cD, 256, 0, stream>>>(desc, qw, qb, desc_mask, qn);
  ln_rows_kernel<<<cB * cL, 256, 0, stream>>>(dna, kvw, kvb, nullptr, kvn);

  // 3) QKV projections (WMMA, 16x64 strip per wave)
  const float scale = 0.125f;                    // 1/sqrt(64)
  int Mq = cB * cD, Mkv = cB * cL;
  int blocksQ  = (Mq >> 4) * (cH >> 6) / 8;      // 1024
  int blocksKV = (Mkv >> 4) * (cH >> 6) / 8;     // 2048
  gemm_wmma_kernel<<<blocksQ, 256, 0, stream>>>(qn, wbf, inb, qp,
                                                Mq, cH, cH, scale, 0, 0);
  gemm_wmma_kernel<<<blocksKV, 256, 0, stream>>>(kvn, wbf + (long)cH * cH, inb + cH, kp,
                                                 Mkv, cH, cH, 1.0f, 1, 0);
  gemm_wmma_kernel<<<blocksKV, 256, 0, stream>>>(kvn, wbf + 2L * cH * cH, inb + 2 * cH, vt,
                                                 Mkv, cH, cH, 1.0f, 2, cL);

  // 4) fused attention (dynamic LDS: 16x2048 f32 scores + reductions)
  int smemBytes = (16 * cL + 256 + 256 + 16 + 16) * (int)sizeof(float);
  (void)hipFuncSetAttribute((const void*)attn_fused_kernel,
                            hipFuncAttributeMaxDynamicSharedMemorySize, smemBytes);
  attn_fused_kernel<<<cB * cNH * (cD / 16), 256, smemBytes, stream>>>(
      qp, kp, vt, dna_mask, attn, ctx);

  // 5) out projection (WMMA)
  gemm_wmma_kernel<<<blocksQ, 256, 0, stream>>>(ctx, wbf + 3L * cH * cH, ob, ao,
                                                Mq, cH, cH, 1.0f, 0, 0);

  // 6) pooled = masked-LN(ao + qn) averaged over valid queries
  zero_pool_count_kernel<<<cB, 256, 0, stream>>>(pooled, desc_mask, cnt);
  final_pool_kernel<<<cB * cD, 256, 0, stream>>>(ao, qn, onw, onb, desc_mask, cnt, pooled);
}
